// ModelAttention2Layers_46789373722916
// MI455X (gfx1250) — matrix operand order repaired and checked by reading, not compiled, not run on hardware
//
#include <hip/hip_runtime.h>
#include <math.h>

// ---------------------------------------------------------------------------
// B=8, S=2048, D=512, K=256 — output is final[0, -1, :]:
//   * attention mixes only within a batch  -> only batch 0 is live (8x cut)
//   * block 3 only needs its last query row (full K3/V3, 1-row Q3/out3)
// All matmuls on v_wmma_f32_16x16x32_f16 (f16 inputs, fp32 accumulate).
// Each wave computes a 16x64 C strip (4 WMMAs / k-step, A fragment reused 4x);
// the 32x64 B chunk is staged once per 128-thread block into LDS as f16 and
// shared by all 4 waves (read back as aligned ds_load_b128 pairs).
// ---------------------------------------------------------------------------

typedef _Float16 half16 __attribute__((ext_vector_type(16)));
typedef _Float16 half8  __attribute__((ext_vector_type(8)));
typedef float    float8 __attribute__((ext_vector_type(8)));

#define SEQ 2048
#define DD  512
#define KK_ 256

#define LDS_STRIDE 40   // halves per LDS column (80 B: keeps b128 reads aligned)

// C[M x N] = A[M x Kd] * B (+ bias)
//   BT=false: B is (Kd x N) row-major, ldb = row stride
//   BT=true : A * B2^T, B2 is (N x Kd) row-major, ldb = row stride of B2
// blockDim.x = 128 (4 waves); block tile 64(M) x 64(N); wave w owns M-tile w.
// N % 64 == 0, Kd % 32 == 0; M arbitrary when GUARD=true (else M % 64 == 0).
template<bool BT, bool HAS_BIAS, bool GUARD>
__global__ void gemm_wmma_f16(const float* __restrict__ A, int lda,
                              const float* __restrict__ B, int ldb,
                              const float* __restrict__ bias,
                              float* __restrict__ C, int ldc,
                              int M, int N, int Kd)
{
    __shared__ __align__(64) _Float16 Bs[64 * LDS_STRIDE];

    const int tid  = threadIdx.x;
    const int lane = tid & 31;
    const int wave = tid >> 5;
    const int l    = lane & 15;           // row (A) / col (B,C) within a tile
    const int hi   = lane >> 4;           // selects K sub-range of the fragment
    const int m0w  = (blockIdx.y << 6) + (wave << 4);   // this wave's M-tile
    const int n0   = blockIdx.x << 6;                   // block's 64-col strip

    int arow = m0w + l;
    if (GUARD && arow >= M) arow = M - 1; // clamped loads; stores are skipped
    const float* __restrict__ Arow = A + (size_t)arow * lda + hi * 8;

    float8 c0 = {}, c1 = {}, c2 = {}, c3 = {};

    for (int kk = 0; kk < Kd; kk += 32) {
        // ---- stage 32(K) x 64(N) B chunk -> LDS f16, Bs[col*STRIDE + kloc]
        __syncthreads();                  // previous iteration's reads done
        if (BT) {
            // B2 rows n0..n0+63, k cols kk..kk+31 : contiguous along k
            #pragma unroll
            for (int j = 0; j < 4; ++j) {
                const int idx  = tid + j * 128;     // 512 float4 total
                const int bcol = idx >> 3;          // 0..63 (local col / B2 row)
                const int kq   = (idx & 7) << 2;    // 0,4,..,28
                const float4 bv = *(const float4*)(B + (size_t)(n0 + bcol) * ldb + kk + kq);
                _Float16* p = &Bs[bcol * LDS_STRIDE + kq];
                p[0] = (_Float16)bv.x; p[1] = (_Float16)bv.y;
                p[2] = (_Float16)bv.z; p[3] = (_Float16)bv.w;
            }
        } else {
            // B rows kk..kk+31, cols n0..n0+63 : contiguous along n (transpose in)
            #pragma unroll
            for (int j = 0; j < 4; ++j) {
                const int idx  = tid + j * 128;     // 512 float4 total
                const int krow = idx >> 4;          // 0..31
                const int c4   = (idx & 15) << 2;   // 0,4,..,60
                const float4 bv = *(const float4*)(B + (size_t)(kk + krow) * ldb + n0 + c4);
                Bs[(c4 + 0) * LDS_STRIDE + krow] = (_Float16)bv.x;
                Bs[(c4 + 1) * LDS_STRIDE + krow] = (_Float16)bv.y;
                Bs[(c4 + 2) * LDS_STRIDE + krow] = (_Float16)bv.z;
                Bs[(c4 + 3) * LDS_STRIDE + krow] = (_Float16)bv.w;
            }
        }
        __syncthreads();

        // ---- A fragment (16x32): lane(l,hi) row `arow`,
        //      K = kk + hi*8 + {0..7} (VGPR0-3) and + {16..23} (VGPR4-7)
        const float4 a0 = *(const float4*)(Arow + kk);
        const float4 a1 = *(const float4*)(Arow + kk + 4);
        const float4 a2 = *(const float4*)(Arow + kk + 16);
        const float4 a3 = *(const float4*)(Arow + kk + 20);
        half16 a;
        a[0]=(_Float16)a0.x;  a[1]=(_Float16)a0.y;  a[2]=(_Float16)a0.z;  a[3]=(_Float16)a0.w;
        a[4]=(_Float16)a1.x;  a[5]=(_Float16)a1.y;  a[6]=(_Float16)a1.z;  a[7]=(_Float16)a1.w;
        a[8]=(_Float16)a2.x;  a[9]=(_Float16)a2.y;  a[10]=(_Float16)a2.z; a[11]=(_Float16)a2.w;
        a[12]=(_Float16)a3.x; a[13]=(_Float16)a3.y; a[14]=(_Float16)a3.z; a[15]=(_Float16)a3.w;

        // ---- 4 B fragments from LDS, 4 WMMAs reusing `a`
        #pragma unroll
        for (int t = 0; t < 4; ++t) {
            const _Float16* bp = &Bs[(t * 16 + l) * LDS_STRIDE + hi * 8];
            const half8 b0 = *(const half8*)(bp);       // 16B-aligned
            const half8 b1 = *(const half8*)(bp + 16);  // 16B-aligned
            const half16 b = __builtin_shufflevector(b0, b1,
                0, 1, 2, 3, 4, 5, 6, 7, 8, 9, 10, 11, 12, 13, 14, 15);
            if      (t == 0) c0 = __builtin_amdgcn_wmma_f32_16x16x32_f16(false, a, false, b, (short)0, c0, false, false);
            else if (t == 1) c1 = __builtin_amdgcn_wmma_f32_16x16x32_f16(false, a, false, b, (short)0, c1, false, false);
            else if (t == 2) c2 = __builtin_amdgcn_wmma_f32_16x16x32_f16(false, a, false, b, (short)0, c2, false, false);
            else             c3 = __builtin_amdgcn_wmma_f32_16x16x32_f16(false, a, false, b, (short)0, c3, false, false);
        }
    }

    // ---- epilogue: C layout VGPR r -> row m0w + r + hi*8, col n0 + t*16 + l
    #pragma unroll
    for (int t = 0; t < 4; ++t) {
        const float8& c = (t == 0) ? c0 : (t == 1) ? c1 : (t == 2) ? c2 : c3;
        const int n = n0 + t * 16 + l;
        const float bv = HAS_BIAS ? bias[n] : 0.0f;
        #pragma unroll
        for (int r = 0; r < 8; ++r) {
            const int m = m0w + r + hi * 8;
            if (!GUARD || m < M) C[(size_t)m * ldc + n] = c[r] + bv;
        }
    }
}

// Row-wise softmax in place (one block per row).
__global__ void softmax_rows(float* __restrict__ S, int cols)
{
    float* __restrict__ p = S + (size_t)blockIdx.x * cols;
    __shared__ float red[256];
    const int t = threadIdx.x;

    float mx = -3.0e38f;
    for (int i = t; i < cols; i += 256) mx = fmaxf(mx, p[i]);
    red[t] = mx; __syncthreads();
    for (int s = 128; s > 0; s >>= 1) { if (t < s) red[t] = fmaxf(red[t], red[t + s]); __syncthreads(); }
    mx = red[0]; __syncthreads();

    float sum = 0.0f;
    for (int i = t; i < cols; i += 256) { float e = __expf(p[i] - mx); p[i] = e; sum += e; }
    red[t] = sum; __syncthreads();
    for (int s = 128; s > 0; s >>= 1) { if (t < s) red[t] += red[t + s]; __syncthreads(); }
    const float inv = 1.0f / red[0];

    for (int i = t; i < cols; i += 256) p[i] *= inv;
}

// Row-wise L2 normalization in place (one block per row).
__global__ void l2norm_rows(float* __restrict__ V, int cols)
{
    float* __restrict__ p = V + (size_t)blockIdx.x * cols;
    __shared__ float red[256];
    const int t = threadIdx.x;

    float ss = 0.0f;
    for (int i = t; i < cols; i += 256) { float v = p[i]; ss += v * v; }
    red[t] = ss; __syncthreads();
    for (int s = 128; s > 0; s >>= 1) { if (t < s) red[t] += red[t + s]; __syncthreads(); }
    const float inv = rsqrtf(red[0]);

    for (int i = t; i < cols; i += 256) p[i] *= inv;
}

// ---------------------------------------------------------------------------

static inline dim3 gemm_grid(int M, int N) { return dim3(N / 64, (M + 63) / 64); }

extern "C" void kernel_launch(void* const* d_in, const int* in_sizes, int n_in,
                              void* d_out, int out_size, void* d_ws, size_t ws_size,
                              hipStream_t stream)
{
    const float* x   = (const float*)d_in[0];   // (8, 2048, 512) — batch 0 only
    const float* Wk1 = (const float*)d_in[1];
    const float* bk1 = (const float*)d_in[2];
    const float* Wq1 = (const float*)d_in[3];
    const float* bq1 = (const float*)d_in[4];
    const float* Wk2 = (const float*)d_in[5];
    const float* bk2 = (const float*)d_in[6];
    const float* Wq2 = (const float*)d_in[7];
    const float* bq2 = (const float*)d_in[8];
    const float* Wv2 = (const float*)d_in[9];
    const float* bv2 = (const float*)d_in[10];
    float* out = (float*)d_out;                 // 512 floats

    // Workspace layout (32 MB of floats)
    float* ws = (float*)d_ws;
    float* Q  = ws;                              // 2048 x 256
    float* Kb = Q  + (size_t)SEQ * KK_;          // 2048 x 256
    float* V  = Kb + (size_t)SEQ * KK_;          // 2048 x 512
    float* Sc = V  + (size_t)SEQ * DD;           // 2048 x 2048 (scores / probs)
    float* H1 = Sc + (size_t)SEQ * SEQ;          // 2048 x 512
    float* H2 = H1 + (size_t)SEQ * DD;           // 2048 x 512

    const dim3 blk(128);

    // ===== Block 1 (values = raw x, batch 0) =====
    gemm_wmma_f16<false, true,  false><<<gemm_grid(SEQ, KK_), blk, 0, stream>>>(x, DD, Wq1, KK_, bq1, Q,  KK_, SEQ, KK_, DD);
    gemm_wmma_f16<false, true,  false><<<gemm_grid(SEQ, KK_), blk, 0, stream>>>(x, DD, Wk1, KK_, bk1, Kb, KK_, SEQ, KK_, DD);
    gemm_wmma_f16<true,  false, false><<<gemm_grid(SEQ, SEQ), blk, 0, stream>>>(Q, KK_, Kb,  KK_, nullptr, Sc, SEQ, SEQ, SEQ, KK_);
    softmax_rows<<<dim3(SEQ), dim3(256), 0, stream>>>(Sc, SEQ);
    gemm_wmma_f16<false, false, false><<<gemm_grid(SEQ, DD),  blk, 0, stream>>>(Sc, SEQ, x,  DD,  nullptr, H1, DD, SEQ, DD, SEQ);

    // ===== Block 2 (H1 -> H2) =====
    gemm_wmma_f16<false, true,  false><<<gemm_grid(SEQ, KK_), blk, 0, stream>>>(H1, DD, Wq2, KK_, bq2, Q,  KK_, SEQ, KK_, DD);
    gemm_wmma_f16<false, true,  false><<<gemm_grid(SEQ, KK_), blk, 0, stream>>>(H1, DD, Wk2, KK_, bk2, Kb, KK_, SEQ, KK_, DD);
    gemm_wmma_f16<false, true,  false><<<gemm_grid(SEQ, DD),  blk, 0, stream>>>(H1, DD, Wv2, DD,  bv2, V,  DD,  SEQ, DD,  DD);
    l2norm_rows<<<dim3(SEQ), dim3(256), 0, stream>>>(V, DD);
    gemm_wmma_f16<true,  false, false><<<gemm_grid(SEQ, SEQ), blk, 0, stream>>>(Q, KK_, Kb,  KK_, nullptr, Sc, SEQ, SEQ, SEQ, KK_);
    softmax_rows<<<dim3(SEQ), dim3(256), 0, stream>>>(Sc, SEQ);
    gemm_wmma_f16<false, false, false><<<gemm_grid(SEQ, DD),  blk, 0, stream>>>(Sc, SEQ, V,  DD,  nullptr, H2, DD, SEQ, DD, SEQ);

    // ===== Block 3 (same weights; only query row 2047 is live) =====
    gemm_wmma_f16<false, true,  false><<<gemm_grid(SEQ, KK_), blk, 0, stream>>>(H2, DD, Wk2, KK_, bk2, Kb, KK_, SEQ, KK_, DD);
    gemm_wmma_f16<false, true,  false><<<gemm_grid(SEQ, DD),  blk, 0, stream>>>(H2, DD, Wv2, DD,  bv2, V,  DD,  SEQ, DD,  DD);
    l2norm_rows<<<dim3(SEQ), dim3(256), 0, stream>>>(V, DD);
    // q for the last row only (M = 1, guarded stores)
    gemm_wmma_f16<false, true,  true ><<<gemm_grid(1, KK_),   blk, 0, stream>>>(H2 + (size_t)(SEQ - 1) * DD, DD,
                                                                                Wq2, KK_, bq2, Q, KK_, 1, KK_, DD);
    gemm_wmma_f16<true,  false, true ><<<gemm_grid(1, SEQ),   blk, 0, stream>>>(Q, KK_, Kb, KK_, nullptr, Sc, SEQ, 1, SEQ, KK_);
    softmax_rows<<<dim3(1), dim3(256), 0, stream>>>(Sc, SEQ);
    // final[0,-1,:] — 1 x 512, written straight to d_out
    gemm_wmma_f16<false, false, true ><<<gemm_grid(1, DD),    blk, 0, stream>>>(Sc, SEQ, V, DD, nullptr, out, DD, 1, DD, SEQ);
}
